// SelfAttention_27324581937382
// MI455X (gfx1250) — matrix operand order repaired and checked by reading, not compiled, not run on hardware
//
#include <hip/hip_runtime.h>
#include <hip/hip_bf16.h>

typedef __bf16 bf16_t;
typedef __attribute__((ext_vector_type(16))) __bf16 v16bf;
typedef __attribute__((ext_vector_type(8)))  __bf16 v8bf;
typedef __attribute__((ext_vector_type(4)))  __bf16 v4bf;
typedef __attribute__((ext_vector_type(8)))  float  v8f;
typedef int v4i_vs __attribute__((vector_size(16)));       // async builtin param type
typedef unsigned int u32x4 __attribute__((vector_size(16)));
typedef int i32x8 __attribute__((vector_size(32)));
typedef int i32x4 __attribute__((vector_size(16)));

enum { OUT_F32 = 0, OUT_BF16 = 1, OUT_BF16_T = 2 };

#ifndef __has_builtin
#define __has_builtin(x) 0
#endif

#if __has_builtin(__builtin_amdgcn_tensor_load_to_lds) && \
    __has_builtin(__builtin_amdgcn_s_wait_tensorcnt)
#define USE_TDM 1
#else
#define USE_TDM 0
#endif

#if __has_builtin(__builtin_amdgcn_global_load_async_to_lds_b128)
#define USE_ASYNC_LDS 1
#else
#define USE_ASYNC_LDS 0
#endif

// LDS byte offset of a __shared__ object (AS3 addresses are LDS offsets).
__device__ __forceinline__ unsigned lds_offset_of(const void* p) {
    return (unsigned)(size_t)(__attribute__((address_space(3))) const char*)p;
}

#if USE_TDM
// One TDM descriptor: load a rows x 32 bf16 tile (row stride ld elements) from
// global into LDS at lds_off, inserting 16B padding after each 64B row so the
// LDS image has row stride 40 bf16 (matches As/Bs layout).
__device__ __forceinline__
void tdm_load_tile(unsigned lds_off, const bf16_t* gaddr, long long ld, int rows) {
    unsigned long long ga = (unsigned long long)(size_t)gaddr;
    u32x4 g0;
    g0[0] = 1u;                                         // count=1 (valid), user mode
    g0[1] = lds_off;                                    // lds_addr (bytes)
    g0[2] = (unsigned)(ga & 0xFFFFFFFFu);               // global_addr[31:0]
    g0[3] = (unsigned)((ga >> 32) & 0x1FFFFFFu)         // global_addr[56:32]
          | (2u << 30);                                 // type=2 ("image")
    const unsigned td0 = 0x7FFFFFFFu, td1 = 0x7FFFFFFFu; // huge dims: no OOB clamp
    const unsigned long long st0 = (unsigned long long)ld;
    i32x8 g1;
    g1[0] = (int)((1u << 16)      // data_size = 1 -> 2 bytes
                | (1u << 20)      // pad_enable
                | (3u << 22)      // pad_interval = 16 DWORDs (one 64B row)
                | (3u << 25));    // pad_amount   = 4 DWORDs (16B)
    g1[1] = (int)((td0 & 0xFFFFu) << 16);               // [47:32]=abar=0, td0 lo16
    g1[2] = (int)(((td0 >> 16) & 0xFFFFu) | ((td1 & 0xFFFFu) << 16));
    g1[3] = (int)(((td1 >> 16) & 0xFFFFu) | (32u << 16)); // tile_dim0 = 32
    g1[4] = (int)(unsigned)rows;                        // tile_dim1 = rows, tile_dim2 = 0
    g1[5] = (int)(st0 & 0xFFFFFFFFu);                   // tensor_dim0_stride lo32
    g1[6] = (int)((st0 >> 32) & 0xFFFFu);               // stride hi16; dim1_stride = 0
    g1[7] = 0;
    const i32x4 z4 = {0, 0, 0, 0};                      // groups 2/3: 2D tile
    const i32x8 z8 = {0, 0, 0, 0, 0, 0, 0, 0};          // extra group (6-arg form)
    __builtin_amdgcn_tensor_load_to_lds(g0, g1, z4, z4, z8, 0);
}
#define WAIT_TILE_PAIR() __builtin_amdgcn_s_wait_tensorcnt(2)
#define WAIT_TILE_NONE() __builtin_amdgcn_s_wait_tensorcnt(0)
#elif USE_ASYNC_LDS
#if __has_builtin(__builtin_amdgcn_s_wait_asynccnt)
#define WAIT_TILE_PAIR() __builtin_amdgcn_s_wait_asynccnt(6)
#define WAIT_TILE_NONE() __builtin_amdgcn_s_wait_asynccnt(0)
#else
#define WAIT_TILE_PAIR() asm volatile("s_wait_asynccnt 6" ::: "memory")
#define WAIT_TILE_NONE() asm volatile("s_wait_asynccnt 0" ::: "memory")
#endif
#else
#define WAIT_TILE_PAIR()
#define WAIT_TILE_NONE()
#endif

// 16-byte global -> LDS copy (per lane), used when TDM is unavailable.
__device__ __forceinline__ void copy16_g2l(const bf16_t* g, bf16_t* l) {
#if USE_ASYNC_LDS
    __builtin_amdgcn_global_load_async_to_lds_b128(
        (__attribute__((address_space(1))) v4i_vs*)g,
        (__attribute__((address_space(3))) v4i_vs*)l,
        0, 0);
#else
    *(v8bf*)l = *(const v8bf*)g;
#endif
}

// ---------------------------------------------------------------------------
// fp32 -> bf16 elementwise convert (vectorized x4)
// ---------------------------------------------------------------------------
__global__ void k_f32_to_bf16(const float* __restrict__ in,
                              bf16_t* __restrict__ out, long long n) {
    long long i = ((long long)blockIdx.x * blockDim.x + threadIdx.x) * 4;
    if (i + 3 < n) {
        float4 f = *(const float4*)(in + i);
        v4bf o;
        o[0] = (bf16_t)f.x; o[1] = (bf16_t)f.y;
        o[2] = (bf16_t)f.z; o[3] = (bf16_t)f.w;
        *(v4bf*)(out + i) = o;
    }
}

// ---------------------------------------------------------------------------
// fp32 (rows x cols) -> bf16 transposed (cols x rows); 32x32 LDS tile
// ---------------------------------------------------------------------------
__global__ __launch_bounds__(256)
void k_f32_to_bf16_T(const float* __restrict__ in, bf16_t* __restrict__ out,
                     int rows, int cols) {
    __shared__ float tile[32][33];
    const int tx = threadIdx.x, ty = threadIdx.y;      // (32, 8)
    const int c0 = blockIdx.x * 32, r0 = blockIdx.y * 32;
#pragma unroll
    for (int i = 0; i < 4; ++i)
        tile[ty + i * 8][tx] = in[(long long)(r0 + ty + i * 8) * cols + c0 + tx];
    __syncthreads();
#pragma unroll
    for (int i = 0; i < 4; ++i)
        out[(long long)(c0 + ty + i * 8) * rows + r0 + tx] =
            (bf16_t)tile[tx][ty + i * 8];
}

// ---------------------------------------------------------------------------
// Fragment builders (ISA 7.12.2 layouts, wave32)
// A 16x32 bf16: lanes 0-15 row M=l, K 0-7 & 16-23 ; lanes 16-31 K 8-15 & 24-31
// ---------------------------------------------------------------------------
__device__ __forceinline__
v16bf load_a_frag(const bf16_t (*As)[40], int mbase, int lane) {
    int lm = lane & 15, sel = lane >> 4;
    const bf16_t* p = &As[mbase + lm][sel * 8];
    v8bf lo = *(const v8bf*)p;          // K = sel*8 .. +7
    v8bf hi = *(const v8bf*)(p + 16);   // K = sel*8+16 .. +7
    return __builtin_shufflevector(lo, hi, 0,1,2,3,4,5,6,7,8,9,10,11,12,13,14,15);
}

// B 32x16 bf16: lane n (0-15) col n K 0-7 & 16-23 ; lane n+16 col n K 8-15 & 24-31
__device__ __forceinline__
v16bf load_b_frag(const bf16_t (*Bs)[40], int nbase, int lane) {
    int ln = lane & 15, half = lane >> 4;
    const bf16_t* p = &Bs[nbase + ln][half * 8];
    v8bf lo = *(const v8bf*)p;
    v8bf hi = *(const v8bf*)(p + 16);
    return __builtin_shufflevector(lo, hi, 0,1,2,3,4,5,6,7,8,9,10,11,12,13,14,15);
}

// ---------------------------------------------------------------------------
// Tiled bf16 WMMA GEMM, always "NT":  C[z] = A[z] (MxK) * B[z]^T, B is (NxK)
// row-major, optional bias over N. Block 256x128, BK=32, 256 thr = 8 waves
// (4 along M x 2 along N), wave tile 64x64 (4x4 WMMA tiles, 16 WMMA per
// K-step vs 16 frag ds_loads). Double-buffered LDS fed by the Tensor Data
// Mover (one TENSOR_LOAD_TO_LDS per tile, issued by wave 0, TENSORcnt),
// falling back to GLOBAL_LOAD_ASYNC_TO_LDS_B128 (ASYNCcnt) per lane.
// Copies for tile it+1 overlap the WMMAs of tile it.
// ---------------------------------------------------------------------------
template <int OUT_MODE>
__global__ __launch_bounds__(256)
void k_gemm_nt(const bf16_t* __restrict__ A, long long strideA, int lda,
               const bf16_t* __restrict__ B, long long strideB, int ldb,
               const float* __restrict__ bias,
               void* __restrict__ C, long long strideC, int ldc,
               int M, int N, int K) {
    __shared__ bf16_t As[2][256][40];
    __shared__ bf16_t Bs[2][128][40];

    const int tid  = threadIdx.x;
    const int lane = tid & 31;
    const int wave = tid >> 5;
    const int waveRow = wave >> 1;   // 0..3  -> 64 rows each
    const int waveCol = wave & 1;    // 0..1  -> 64 cols each

    const int bm0 = blockIdx.x * 256;
    const int bn0 = blockIdx.y * 128;
    const int z   = blockIdx.z;

    const bf16_t* gA = A + (long long)z * strideA + (long long)bm0 * lda;
    const bf16_t* gB = B + (long long)z * strideB + (long long)bn0 * ldb;

    // Issue the copy of one (A,B) tile pair into LDS buffer `buf`.
    auto issue_tiles = [&](int buf, int ko) {
#if USE_TDM
        if (wave == 0) {
            tdm_load_tile(lds_offset_of(&As[buf][0][0]), gA + ko, lda, 256);
            tdm_load_tile(lds_offset_of(&Bs[buf][0][0]), gB + ko, ldb, 128);
        }
#else
        // A: 1024 segs of 8 bf16 (256x32), 4 per thread.
#pragma unroll
        for (int i = 0; i < 4; ++i) {
            int s = tid + i * 256;
            int r = s >> 2, k = (s & 3) * 8;
            const bf16_t* pa = gA + (long long)r * lda + ko + k;
            copy16_g2l(pa, &As[buf][r][k]);
            __builtin_prefetch((const void*)(pa + 32), 0, 1);
        }
        // B: 512 segs (128x32), 2 per thread.
#pragma unroll
        for (int i = 0; i < 2; ++i) {
            int s = tid + i * 256;
            int r = s >> 2, k = (s & 3) * 8;
            const bf16_t* pb = gB + (long long)r * ldb + ko + k;
            copy16_g2l(pb, &Bs[buf][r][k]);
            __builtin_prefetch((const void*)(pb + 32), 0, 1);
        }
#endif
    };

    issue_tiles(0, 0);               // tile 0 in flight

    v8f acc[4][4];
#pragma unroll
    for (int f = 0; f < 4; ++f)
#pragma unroll
        for (int g = 0; g < 4; ++g)
#pragma unroll
            for (int v = 0; v < 8; ++v) acc[f][g][v] = 0.0f;

    const int nkt = K >> 5;
    for (int it = 0; it < nkt; ++it) {
        const int cur = it & 1;
        // End-of-previous-step barrier guarantees no wave still reads cur^1,
        // so issuing the next copies into it is WAR-safe.
        if (it + 1 < nkt) {
            issue_tiles(cur ^ 1, (it + 1) * 32);
            WAIT_TILE_PAIR();        // oldest pair (tile it) complete, in order
        } else {
            WAIT_TILE_NONE();
        }
        __syncthreads();             // tile `it` visible to all waves

        // All 4 A frags, then per-B-frag issue 4 WMMAs (caps frag liveness).
        v16bf af[4];
#pragma unroll
        for (int f = 0; f < 4; ++f)
            af[f] = load_a_frag(As[cur], waveRow * 64 + f * 16, lane);
#pragma unroll
        for (int g = 0; g < 4; ++g) {
            v16bf bfr = load_b_frag(Bs[cur], waveCol * 64 + g * 16, lane);
#pragma unroll
            for (int f = 0; f < 4; ++f)
                acc[f][g] = __builtin_amdgcn_wmma_f32_16x16x32_bf16(
                    false, af[f], false, bfr,
                    (short)0, acc[f][g], false, false);
        }

        __syncthreads();             // protect buf `cur` before next issue
    }

    // Epilogue: C/D layout — VGPR v: M = v + 8*(lane>>4), N = lane&15
    const int colLane = lane & 15;
    const int rowHalf = (lane >> 4) * 8;
#pragma unroll
    for (int f = 0; f < 4; ++f) {
#pragma unroll
        for (int g = 0; g < 4; ++g) {
            const int col  = bn0 + waveCol * 64 + g * 16 + colLane;
            const int row0 = bm0 + waveRow * 64 + f * 16 + rowHalf;
            const float bv = bias ? bias[col] : 0.0f;
            if constexpr (OUT_MODE == OUT_BF16) {
                bf16_t* Cb = (bf16_t*)C + (long long)z * strideC;
#pragma unroll
                for (int v = 0; v < 8; ++v)
                    Cb[(long long)(row0 + v) * ldc + col] = (bf16_t)(acc[f][g][v] + bv);
            } else if constexpr (OUT_MODE == OUT_BF16_T) {
                // Transposed store: lane's 8 M-values are contiguous in C^T.
                bf16_t* Ct = (bf16_t*)C + (long long)z * strideC;
                v8bf pk;
#pragma unroll
                for (int v = 0; v < 8; ++v) pk[v] = (bf16_t)(acc[f][g][v] + bv);
                *(v8bf*)(Ct + (long long)col * ldc + row0) = pk;
            } else {
                float* Cf = (float*)C + (long long)z * strideC;
#pragma unroll
                for (int v = 0; v < 8; ++v)
                    Cf[(long long)(row0 + v) * ldc + col] = acc[f][g][v] + bv;
            }
        }
    }
}

// ---------------------------------------------------------------------------
// Row softmax over 2048 fp32 scores with scale 1/32, writing bf16 probs
// IN PLACE over the fp32 row (bf16 row occupies first half of the fp32 row).
// One block (256 threads) per row; 8 values per thread.
// ---------------------------------------------------------------------------
__global__ __launch_bounds__(256)
void k_softmax_rows(float* __restrict__ S, int ncols) {
    const int t = threadIdx.x;
    float* rp = S + (long long)blockIdx.x * ncols;
    const float scale = 0.03125f;   // 1/sqrt(1024)

    float vals[8];
#pragma unroll
    for (int i = 0; i < 8; ++i) vals[i] = rp[t + i * 256] * scale;   // all reads here

    __shared__ float red[256];
    float m = vals[0];
#pragma unroll
    for (int i = 1; i < 8; ++i) m = fmaxf(m, vals[i]);
    red[t] = m;
    __syncthreads();
    for (int off = 128; off > 0; off >>= 1) {
        if (t < off) red[t] = fmaxf(red[t], red[t + off]);
        __syncthreads();
    }
    m = red[0];
    __syncthreads();

    float p[8];
    float s = 0.0f;
#pragma unroll
    for (int i = 0; i < 8; ++i) { p[i] = __expf(vals[i] - m); s += p[i]; }
    red[t] = s;
    __syncthreads();
    for (int off = 128; off > 0; off >>= 1) {
        if (t < off) red[t] += red[t + off];
        __syncthreads();
    }
    const float inv = 1.0f / red[0];

    bf16_t* op = (bf16_t*)rp;       // in-place: bf16 row over fp32 row
#pragma unroll
    for (int i = 0; i < 8; ++i) op[t + i * 256] = (bf16_t)(p[i] * inv);
}

// ---------------------------------------------------------------------------
// Host-side orchestration
// ---------------------------------------------------------------------------
extern "C" void kernel_launch(void* const* d_in, const int* in_sizes, int n_in,
                              void* d_out, int out_size, void* d_ws, size_t ws_size,
                              hipStream_t stream) {
    (void)in_sizes; (void)n_in; (void)out_size; (void)ws_size;

    const int B = 4, S = 2048, D = 1024;
    const long long MT = (long long)B * S;          // 8192 token rows

    const float* x  = (const float*)d_in[0];
    const float* Wq = (const float*)d_in[1];
    const float* bq = (const float*)d_in[2];
    const float* Wk = (const float*)d_in[3];
    const float* bk = (const float*)d_in[4];
    const float* Wv = (const float*)d_in[5];
    const float* bv = (const float*)d_in[6];
    float* out = (float*)d_out;

    // ---- workspace layout ----
    char* w = (char*)d_ws;
    bf16_t* xb  = (bf16_t*)w; w += MT * D * sizeof(bf16_t);             // 16 MB
    bf16_t* Wqt = (bf16_t*)w; w += (long long)D * D * sizeof(bf16_t);   // 2 MB (W^T)
    bf16_t* Wkt = (bf16_t*)w; w += (long long)D * D * sizeof(bf16_t);
    bf16_t* Wvt = (bf16_t*)w; w += (long long)D * D * sizeof(bf16_t);
    bf16_t* Qb  = (bf16_t*)w; w += MT * D * sizeof(bf16_t);             // 16 MB
    bf16_t* Kb  = (bf16_t*)w; w += MT * D * sizeof(bf16_t);             // 16 MB
    bf16_t* Vt  = (bf16_t*)w; w += MT * D * sizeof(bf16_t);             // 16 MB (V^T: D x MT)
    float*  Sc  = (float*)w;                                            // 64 MB
    // total ~134 MB of d_ws

    // ---- 1) converts: x plain; weights transposed to (N x K) ----
    {
        long long nx = MT * D;
        k_f32_to_bf16<<<dim3((unsigned)(nx / 1024)), dim3(256), 0, stream>>>(x, xb, nx);
        dim3 tg(D / 32, D / 32), tb(32, 8);
        k_f32_to_bf16_T<<<tg, tb, 0, stream>>>(Wq, Wqt, D, D);
        k_f32_to_bf16_T<<<tg, tb, 0, stream>>>(Wk, Wkt, D, D);
        k_f32_to_bf16_T<<<tg, tb, 0, stream>>>(Wv, Wvt, D, D);
    }

    // ---- 2) projections (NT with W^T): Q, K normal bf16; V stored transposed ----
    {
        dim3 grid(32, 8, 1), blk(256);
        k_gemm_nt<OUT_BF16><<<grid, blk, 0, stream>>>(
            xb, 0, D, Wqt, 0, D, bq, (void*)Qb, 0, D, (int)MT, D, D);
        k_gemm_nt<OUT_BF16><<<grid, blk, 0, stream>>>(
            xb, 0, D, Wkt, 0, D, bk, (void*)Kb, 0, D, (int)MT, D, D);
        k_gemm_nt<OUT_BF16_T><<<grid, blk, 0, stream>>>(
            xb, 0, D, Wvt, 0, D, bv, (void*)Vt, 0, (int)MT, (int)MT, D, D);
    }

    // ---- 3) scores: Sc[z] = Q[z] * K[z]^T  (2048x2048 f32, batched) ----
    {
        dim3 grid(8, 16, 4), blk(256);
        k_gemm_nt<OUT_F32><<<grid, blk, 0, stream>>>(
            Qb, (long long)S * D, D,
            Kb, (long long)S * D, D,
            nullptr,
            (void*)Sc, (long long)S * S, S,
            S, S, D);
    }

    // ---- 4) softmax rows (scale 1/32), bf16 probs in place over Sc ----
    k_softmax_rows<<<dim3((unsigned)(B * S)), dim3(256), 0, stream>>>(Sc, S);

    // ---- 5) out[z] = P[z] * V[z] ; B operand is V^T rows (NT) ----
    {
        const bf16_t* P = (const bf16_t*)Sc;
        dim3 grid(8, 8, 4), blk(256);
        k_gemm_nt<OUT_F32><<<grid, blk, 0, stream>>>(
            P, (long long)S * (2 * S), 2 * S,      // bf16 row stride = 4096 elems
            Vt, (long long)S, (int)MT,             // B_z(n,k) = Vt[n][z*S + k]
            nullptr,
            (void*)out, (long long)S * D, D,
            S, D, S);
    }
}
